// SubgraphClassficationModel_71966472012459
// MI455X (gfx1250) — compile-verified, matrix operand order checked by reading, not run on hardware
//
#include <hip/hip_runtime.h>
#include <hip/hip_bf16.h>

typedef _Float16 f16;
typedef __attribute__((ext_vector_type(16))) _Float16 v16h;
typedef __attribute__((ext_vector_type(8)))  _Float16 v8h;
typedef __attribute__((ext_vector_type(8)))  float    v8f;
typedef __attribute__((ext_vector_type(4)))  float    v4f;

// ---------------- flat param offsets (jax tree_leaves: sorted dict keys) ----
static constexpr long P_WE_WH    = 0;        // (1,1)
static constexpr long P_WE_WSB   = 1;        // (32,)
static constexpr long P_WE_WSW   = 33;       // (32,33)
static constexpr long P_WE_WV    = 1089;     // (1,1)
static constexpr long P_WELN_B   = 1090;     // (32,)
static constexpr long P_WELN_G   = 1122;     // (32,)
static constexpr long P_WOUT_WH  = 1154;     // (16,16)
static constexpr long P_WOUT_WSB = 1410;     // (100,)
static constexpr long P_WOUT_WSW = 1510;     // (100,116)
static constexpr long P_WOUTLN_B = 13110;    // (100,)
static constexpr long P_WOUTLN_G = 13210;    // (100,)
static constexpr long P_WV_WH    = 13310;    // (16,3)
static constexpr long P_WV_WSB   = 13358;    // (100,)
static constexpr long P_WV_WSW   = 13458;    // (100,22)
static constexpr long P_WV_WV    = 15658;    // (16,16)
static constexpr long P_WVLN_B   = 15914;    // (6,)
static constexpr long P_WVLN_G   = 15920;    // (6,)
static constexpr long P_D1B      = 15926;    // (200,)
static constexpr long P_D1W      = 16126;    // (200,200)
static constexpr long P_D2B      = 56126;    // (1,)
static constexpr long P_D2W      = 56127;    // (1,200)
static constexpr long P_L0       = 56327;
static constexpr long P_LSTRIDE  = 152613;
static constexpr long L_M0_WH  = 0,      L_M0_WSB = 1089,   L_M0_WSW = 1189,   L_M0_WV = 27689;
static constexpr long L_M1_WH  = 28217,  L_M1_WSB = 28473,  L_M1_WSW = 28573,  L_M1_WV = 40173;
static constexpr long L_M2_WH  = 40429,  L_M2_WSB = 40685,  L_M2_WSW = 40785,  L_M2_WV = 52385;
static constexpr long L_FF0_WH = 52641,  L_FF0_WSB= 53153,  L_FF0_WSW= 53553,  L_FF0_WV= 106353;
static constexpr long L_FF1_WH = 107377, L_FF1_WSB= 108401, L_FF1_WSW= 108501, L_FF1_WV= 151701;
static constexpr long L_N0_B = 152213, L_N0_G = 152313, L_N1_B = 152413, L_N1_G = 152513;

// ---------------- packed (padded f16) weight offsets, in halfs --------------
static constexpr long WP_WV   = 0;                 // (112,32)
static constexpr long WP_WE   = 3584;              // (32,64)
static constexpr long WP_WOUT = 5632;              // (112,128)
static constexpr long WP_L0   = 19968;
static constexpr long WP_LSTRIDE = 175104;
static constexpr long WL_M0 = 0;        // (112,288)
static constexpr long WL_M1 = 32256;    // (112,128)
static constexpr long WL_M2 = 46592;    // (112,128)
static constexpr long WL_FF0 = 60928;   // (400,160)
static constexpr long WL_FF1 = 124928;  // (112,448)
static constexpr long WP_TOTAL = WP_L0 + 4 * WP_LSTRIDE;  // 720384 halfs
// padded f32 bias offsets, in floats
static constexpr long BP_WV = 0, BP_WE = 112, BP_WOUT = 144, BP_L0 = 256, BP_LSTRIDE = 848;
static constexpr long BL_M0 = 0, BL_M1 = 112, BL_M2 = 224, BL_FF0 = 336, BL_FF1 = 736;
static constexpr long BP_TOTAL = BP_L0 + 4 * BP_LSTRIDE;  // 3648 floats

static inline long cdiv(long a, long b) { return (a + b - 1) / b; }

// ---------------- utility kernels -------------------------------------------
__global__ void fill_kernel(float* __restrict__ p, float v, long n) {
  long i = (long)blockIdx.x * blockDim.x + threadIdx.x;
  if (i < n) p[i] = v;
}

__global__ void copy_block_kernel(const float* __restrict__ in, long ldi,
                                  float* __restrict__ out, long ldo,
                                  long rows, int cols) {
  long t = (long)blockIdx.x * blockDim.x + threadIdx.x;
  long r = t / cols; int c = (int)(t % cols);
  if (r >= rows) return;
  out[r * ldo + c] = in[r * ldi + c];
}

// pack f32 (Nout x K) into zero-padded f16 (Np x Kp); branchless (clamped load)
__global__ void pack_w_kernel(const float* __restrict__ W, int Nout, int K,
                              f16* __restrict__ out, int Kp, long total) {
  long t = (long)blockIdx.x * blockDim.x + threadIdx.x;
  if (t >= total) return;
  long n = t / Kp; int k = (int)(t % Kp);
  long ncl = (n < Nout) ? n : (Nout - 1);
  int  kcl = (k < K) ? k : (K - 1);
  float v = W[ncl * (long)K + kcl];
  bool ok = (n < (long)Nout) && (k < K);
  out[t] = (f16)(ok ? v : 0.0f);
}

__global__ void pack_b_kernel(const float* __restrict__ B, int Nout,
                              float* __restrict__ out, int Np) {
  int t = blockIdx.x * blockDim.x + threadIdx.x;
  if (t >= Np) return;
  float v = B[(t < Nout) ? t : (Nout - 1)];
  out[t] = (t < Nout) ? v : 0.0f;
}

// s_out = LN(s_in + ds/max(cnt,1)); v_out = (v_in + dv/max(cnt,1)) / sqrt(mean_h max(|v_h|^2,eps))
__global__ void add_ln_kernel(const float* __restrict__ s_in, long ld_si,
                              const float* __restrict__ ds, long ld_ds,
                              const float* __restrict__ cnt,
                              const float* __restrict__ g, const float* __restrict__ b,
                              float* __restrict__ s_out, long ld_so, int D,
                              const float* __restrict__ v_in, const float* __restrict__ dv,
                              float* __restrict__ v_out, int VH, long N) {
  long n = (long)blockIdx.x * blockDim.x + threadIdx.x;
  if (n >= N) return;
  float inv = 1.0f;
  if (cnt) inv = 1.0f / fmaxf(cnt[n], 1.0f);
  const float* sp = s_in + n * ld_si;
  const float* dp = ds ? ds + n * ld_ds : nullptr;
  float mu = 0.0f;
  for (int j = 0; j < D; ++j) mu += sp[j] + (dp ? dp[j] * inv : 0.0f);
  mu /= (float)D;
  float var = 0.0f;
  for (int j = 0; j < D; ++j) {
    float x = sp[j] + (dp ? dp[j] * inv : 0.0f) - mu; var += x * x;
  }
  var /= (float)D;
  float rstd = rsqrtf(var + 1e-5f);
  float* so = s_out + n * ld_so;
  for (int j = 0; j < D; ++j) {
    float x = sp[j] + (dp ? dp[j] * inv : 0.0f);
    so[j] = (x - mu) * rstd * g[j] + b[j];
  }
  if (VH > 0) {
    const float* vp  = v_in + n * (long)VH * 3;
    const float* dvp = dv ? dv + n * (long)VH * 3 : nullptr;
    float m = 0.0f;
    for (int h = 0; h < VH; ++h) {
      float ssq = 0.0f;
      for (int c = 0; c < 3; ++c) {
        float x = vp[h * 3 + c] + (dvp ? dvp[h * 3 + c] * inv : 0.0f);
        ssq += x * x;
      }
      m += fmaxf(ssq, 1e-8f);
    }
    m = sqrtf(m / (float)VH);
    float* vo = v_out + n * (long)VH * 3;
    for (int h = 0; h < VH; ++h)
      for (int c = 0; c < 3; ++c) {
        float x = vp[h * 3 + c] + (dvp ? dvp[h * 3 + c] * inv : 0.0f);
        vo[h * 3 + c] = x / m;
      }
  }
}

// vh[n,h,:] = sum_i wh[h,i]*v[n,i,:]; |vh| written into s_concat[n, off+h]
__global__ void vh_vn_kernel(const float* __restrict__ v, int vi,
                             const float* __restrict__ wh, int H,
                             float* __restrict__ vh,
                             float* __restrict__ s_concat, long lds, int off,
                             long rows) {
  long t = (long)blockIdx.x * blockDim.x + threadIdx.x;
  long n = t / H; int h = (int)(t % H);
  if (n >= rows) return;
  const float* vp = v + n * (long)vi * 3;
  const float* w  = wh + (long)h * vi;
  float a0 = 0.f, a1 = 0.f, a2 = 0.f;
  for (int i = 0; i < vi; ++i) {
    float wv = w[i];
    a0 += wv * vp[i * 3 + 0];
    a1 += wv * vp[i * 3 + 1];
    a2 += wv * vp[i * 3 + 2];
  }
  float* o = vh + (n * (long)H + h) * 3;
  o[0] = a0; o[1] = a1; o[2] = a2;
  s_concat[n * lds + off + h] = sqrtf(fmaxf(a0 * a0 + a1 * a1 + a2 * a2, 1e-8f));
}

// vo[n,o,:] = sum_h wv[o,h]*vh[n,h,:]; optional sigmoid(|vo|) gate
__global__ void wv_gate_kernel(const float* __restrict__ vh, int H,
                               const float* __restrict__ wv, int VO,
                               float* __restrict__ vo, int gate, long rows) {
  long t = (long)blockIdx.x * blockDim.x + threadIdx.x;
  long n = t / VO; int o = (int)(t % VO);
  if (n >= rows) return;
  const float* vp = vh + n * (long)H * 3;
  const float* w  = wv + (long)o * H;
  float a0 = 0.f, a1 = 0.f, a2 = 0.f;
  for (int h = 0; h < H; ++h) {
    float c = w[h];
    a0 += c * vp[h * 3 + 0];
    a1 += c * vp[h * 3 + 1];
    a2 += c * vp[h * 3 + 2];
  }
  if (gate) {
    float nrm = sqrtf(fmaxf(a0 * a0 + a1 * a1 + a2 * a2, 1e-8f));
    float gsc = 1.0f / (1.0f + expf(-nrm));
    a0 *= gsc; a1 *= gsc; a2 *= gsc;
  }
  float* op = vo + (n * (long)VO + o) * 3;
  op[0] = a0; op[1] = a1; op[2] = a2;
}

// ms[e,0:232]=[s[src],es,s[dst]], zero pad [265,288); mv[e]=[v[src],ev,v[dst]]
__global__ void edge_gather_kernel(const float* __restrict__ nS, const float* __restrict__ nV,
                                   const float* __restrict__ eS, const float* __restrict__ eV,
                                   const int* __restrict__ ei, long Etot, long e0, long ce,
                                   float* __restrict__ ms, long ldm,
                                   float* __restrict__ mv) {
  long i = (long)blockIdx.x * blockDim.x + threadIdx.x;
  if (i >= ce) return;
  long e = e0 + i;
  long s = ei[e], d = ei[Etot + e];
  float* m = ms + i * ldm;
  const float* ss = nS + s * 100;
  const float* sd = nS + d * 100;
  for (int j = 0; j < 100; ++j) m[j] = ss[j];
  for (int j = 0; j < 32; ++j)  m[100 + j] = eS[e * 32 + j];
  for (int j = 0; j < 100; ++j) m[132 + j] = sd[j];
  for (int j = 265; j < 288; ++j) m[j] = 0.0f;   // K-pad for WMMA
  float* mm = mv + i * 99;
  const float* vs = nV + s * 48;
  const float* vd = nV + d * 48;
  for (int j = 0; j < 48; ++j) mm[j] = vs[j];
  for (int j = 0; j < 3; ++j)  mm[48 + j] = eV[e * 3 + j];
  for (int j = 0; j < 48; ++j) mm[51 + j] = vd[j];
}

__global__ void scatter_add_kernel(const float* __restrict__ src, long lds, int cols,
                                   const int* __restrict__ idx,
                                   float* __restrict__ dst, long E) {
  long t = (long)blockIdx.x * blockDim.x + threadIdx.x;
  long e = t / cols; int j = (int)(t % cols);
  if (e >= E) return;
  atomicAdd(&dst[(long)idx[e] * cols + j], src[e * lds + j]);
}

__global__ void count_kernel(const int* __restrict__ idx, float* __restrict__ cnt, long E) {
  long e = (long)blockIdx.x * blockDim.x + threadIdx.x;
  if (e >= E) return;
  atomicAdd(&cnt[idx[e]], 1.0f);
}

// ---------------- WMMA GEMM -------------------------------------------------
// out[m,n] = act(sum_k A[m,k]*Wp[n,k] + bias[n])
// Requirements (guaranteed by caller): M % 16 == 0; lda % 32 == 0; K % 32 == 0;
// A pad columns are zero; Wp is zero-padded f16 (Np x K), Np = 16*gridDim.y;
// bias padded to Np. Only the store is guarded by n < N.
__global__ __launch_bounds__(32)
void gemm_wmma_kernel(const float* __restrict__ A, long lda,
                      const f16* __restrict__ Wp,
                      const float* __restrict__ bias,
                      float* __restrict__ out, long ldo,
                      long M, int N, int K, int act) {
  long tileM = (long)blockIdx.x * 16;
  int  tileN = blockIdx.y * 16;
  int lane = threadIdx.x;
  int half = (lane >> 4) & 1;
  int l15  = lane & 15;
  long row = tileM + l15;
  int  col = tileN + l15;
  const float* ap = A + row * lda + half * 8;            // two 8-elem runs/lane
  const f16*   bp = Wp + (long)col * K + half * 8;
  float bi = bias[col];
  v8f acc = {};
  for (int k0 = 0; k0 < K; k0 += 32) {
    v4f a0 = *(const v4f*)(ap + k0);
    v4f a1 = *(const v4f*)(ap + k0 + 4);
    v4f a2 = *(const v4f*)(ap + k0 + 16);
    v4f a3 = *(const v4f*)(ap + k0 + 20);
    v8h b0 = *(const v8h*)(bp + k0);
    v8h b1 = *(const v8h*)(bp + k0 + 16);
    v16h a, b;
#pragma unroll
    for (int j = 0; j < 4; ++j) {
      a[j]      = (f16)a0[j];
      a[4 + j]  = (f16)a1[j];
      a[8 + j]  = (f16)a2[j];
      a[12 + j] = (f16)a3[j];
    }
#pragma unroll
    for (int j = 0; j < 8; ++j) { b[j] = b0[j]; b[8 + j] = b1[j]; }
    acc = __builtin_amdgcn_wmma_f32_16x16x32_f16(false, a, false, b,
                                                 (short)0, acc, false, false);
  }
  if (col < N) {
#pragma unroll
    for (int r = 0; r < 8; ++r) {
      long m = tileM + r + (half ? 8 : 0);
      float v = acc[r] + bi;
      if (act == 1) v = fmaxf(v, 0.0f);
      out[m * ldo + col] = v;
    }
  }
}

// ---------------- head: pooled -> dense(200) relu -> dense(1) -> BCE --------
__global__ void head_kernel(const float* __restrict__ pp, const float* __restrict__ cp,
                            const float* __restrict__ ps, const float* __restrict__ cs,
                            const float* __restrict__ prm,
                            const int* __restrict__ labels,
                            float* __restrict__ out) {
  __shared__ float lg_sm[16];
  int g = threadIdx.x;
  if (g < 16) {
    float x[200];
    float icp = 1.0f / fmaxf(cp[g], 1.0f);
    float ics = 1.0f / fmaxf(cs[g], 1.0f);
    for (int j = 0; j < 100; ++j) x[j]       = pp[g * 100 + j] * icp;
    for (int j = 0; j < 100; ++j) x[100 + j] = ps[g * 100 + j] * ics;
    const float* w1 = prm + P_D1W;
    const float* b1 = prm + P_D1B;
    const float* w2 = prm + P_D2W;
    float lg = prm[P_D2B];
    for (int o = 0; o < 200; ++o) {
      float a = b1[o];
      for (int j = 0; j < 200; ++j) a += w1[(long)o * 200 + j] * x[j];
      a = fmaxf(a, 0.0f);
      lg += w2[o] * a;
    }
    lg_sm[g] = lg;
    out[1 + g] = lg;
  }
  __syncthreads();
  if (g == 0) {
    float loss = 0.0f;
    for (int i = 0; i < 16; ++i) {
      float xv = lg_sm[i];
      float y = (float)labels[i];
      loss += fmaxf(xv, 0.0f) - xv * y + log1pf(expf(-fabsf(xv)));
    }
    out[0] = loss / 16.0f;
  }
}

// ---------------- host-side encoder driver ----------------------------------
static void encode(const float* sV, const float* vV, const int* ei,
                   const float* sE, const float* vE, const int* batch,
                   long N, long E, const float* prm,
                   const f16* wp, const float* bp,
                   float* arena, float* pooled, float* gcnt, hipStream_t st) {
  const long EC = 16384;  // edge chunk (E is always a multiple of EC)
  float* p = arena;
  auto alloc = [&](long n) { float* r = p; p += ((n + 63) & ~63L); return r; };
  float* nS   = alloc(N * 100);
  float* nV   = alloc(N * 48);
  float* eSb  = alloc(E * 32);
  float* eVb  = alloc(E * 3);
  float* big0 = alloc(EC * 288);   // edge concat (ld 288/64) / m2 scalar out (ld 100)
  float* big1 = alloc(EC * 99);    // mv / m2 vector out
  float* big2 = alloc(EC * 99);    // vh
  float* eMSa = alloc(EC * 128);
  float* eMVa = alloc(EC * 48);
  float* eMSb = alloc(EC * 128);
  float* eMVb = alloc(EC * 48);
  float* dS   = alloc(N * 100);
  float* dV   = alloc(N * 48);
  float* cnt  = alloc(N);
  float* nT1  = alloc(N * 448);
  float* nT2  = alloc(N * 160);
  float* nVh  = alloc(N * 96);
  float* nVt  = alloc(N * 96);

  auto l1d = [&](long n) { return dim3((unsigned)cdiv(n, 256)); };
  auto gemm = [&](const float* A, long lda, const f16* W, const float* bias,
                  float* out, long ldo, long M, int Nout, int Kp, int act) {
    dim3 g((unsigned)cdiv(M, 16), (unsigned)cdiv(Nout, 16));
    gemm_wmma_kernel<<<g, dim3(32), 0, st>>>(A, lda, W, bias, out, ldo, M, Nout, Kp, act);
  };

  fill_kernel<<<l1d(1600), 256, 0, st>>>(pooled, 0.0f, 1600);
  fill_kernel<<<l1d(16), 256, 0, st>>>(gcnt, 0.0f, 16);

  // ---- node embedding: Wv_ln + Wv (6,3) -> (100,16); concat ld 32 (K=32)
  fill_kernel<<<l1d(N * 32), 256, 0, st>>>(nT2, 0.0f, N * 32);
  add_ln_kernel<<<l1d(N), 256, 0, st>>>(sV, 6, nullptr, 0, nullptr,
      prm + P_WVLN_G, prm + P_WVLN_B, nT2, 32, 6, vV, nullptr, nVt, 3, N);
  vh_vn_kernel<<<l1d(N * 16), 256, 0, st>>>(nVt, 3, prm + P_WV_WH, 16, nVh, nT2, 32, 6, N);
  gemm(nT2, 32, wp + WP_WV, bp + BP_WV, nS, 100, N, 100, 32, 0);
  wv_gate_kernel<<<l1d(N * 16), 256, 0, st>>>(nVh, 16, prm + P_WV_WV, 16, nV, 0, N);

  // ---- edge embedding: We_ln + We (32,1) -> (32,1); concat ld 64 (K=64)
  for (long e0 = 0; e0 < E; e0 += EC) {
    long ce = (E - e0 < EC) ? (E - e0) : EC;
    fill_kernel<<<l1d(ce * 64), 256, 0, st>>>(big0, 0.0f, ce * 64);
    add_ln_kernel<<<l1d(ce), 256, 0, st>>>(sE + e0 * 32, 32, nullptr, 0, nullptr,
        prm + P_WELN_G, prm + P_WELN_B, big0, 64, 32, vE + e0 * 3, nullptr, eVb + e0 * 3, 1, ce);
    vh_vn_kernel<<<l1d(ce), 256, 0, st>>>(eVb + e0 * 3, 1, prm + P_WE_WH, 1, big1, big0, 64, 32, ce);
    gemm(big0, 64, wp + WP_WE, bp + BP_WE, eSb + e0 * 32, 32, ce, 32, 64, 0);
    wv_gate_kernel<<<l1d(ce), 256, 0, st>>>(big1, 1, prm + P_WE_WV, 1, eVb + e0 * 3, 0, ce);
  }

  // ---- 4 conv layers
  for (int l = 0; l < 4; ++l) {
    const float* lo = prm + P_L0 + (long)l * P_LSTRIDE;
    const f16* wl = wp + WP_L0 + (long)l * WP_LSTRIDE;
    const float* bl = bp + BP_L0 + (long)l * BP_LSTRIDE;
    fill_kernel<<<l1d(N * 100), 256, 0, st>>>(dS, 0.0f, N * 100);
    fill_kernel<<<l1d(N * 48), 256, 0, st>>>(dV, 0.0f, N * 48);
    fill_kernel<<<l1d(N), 256, 0, st>>>(cnt, 0.0f, N);
    for (long e0 = 0; e0 < E; e0 += EC) {
      long ce = (E - e0 < EC) ? (E - e0) : EC;
      fill_kernel<<<l1d(ce * 128), 256, 0, st>>>(eMSa, 0.0f, ce * 128);
      fill_kernel<<<l1d(ce * 128), 256, 0, st>>>(eMSb, 0.0f, ce * 128);
      edge_gather_kernel<<<l1d(ce), 256, 0, st>>>(nS, nV, eSb, eVb, ei, E, e0, ce, big0, 288, big1);
      // m0: (232,33) -> (100,16), relu + sigmoid gate; K=288
      vh_vn_kernel<<<l1d(ce * 33), 256, 0, st>>>(big1, 33, lo + L_M0_WH, 33, big2, big0, 288, 232, ce);
      gemm(big0, 288, wl + WL_M0, bl + BL_M0, eMSa, 128, ce, 100, 288, 1);
      wv_gate_kernel<<<l1d(ce * 16), 256, 0, st>>>(big2, 33, lo + L_M0_WV, 16, eMVa, 1, ce);
      // m1: relu + sigmoid; K=128
      vh_vn_kernel<<<l1d(ce * 16), 256, 0, st>>>(eMVa, 16, lo + L_M1_WH, 16, big2, eMSa, 128, 100, ce);
      gemm(eMSa, 128, wl + WL_M1, bl + BL_M1, eMSb, 128, ce, 100, 128, 1);
      wv_gate_kernel<<<l1d(ce * 16), 256, 0, st>>>(big2, 16, lo + L_M1_WV, 16, eMVb, 1, ce);
      // m2: no acts; K=128
      vh_vn_kernel<<<l1d(ce * 16), 256, 0, st>>>(eMVb, 16, lo + L_M2_WH, 16, big2, eMSb, 128, 100, ce);
      gemm(eMSb, 128, wl + WL_M2, bl + BL_M2, big0, 100, ce, 100, 128, 0);
      wv_gate_kernel<<<l1d(ce * 16), 256, 0, st>>>(big2, 16, lo + L_M2_WV, 16, big1, 0, ce);
      // aggregate to dst
      scatter_add_kernel<<<l1d(ce * 100), 256, 0, st>>>(big0, 100, 100, ei + E + e0, dS, ce);
      scatter_add_kernel<<<l1d(ce * 48), 256, 0, st>>>(big1, 48, 48, ei + E + e0, dV, ce);
      count_kernel<<<l1d(ce), 256, 0, st>>>(ei + E + e0, cnt, ce);
    }
    // residual + norm0 (mean aggregation: divide by cnt)
    add_ln_kernel<<<l1d(N), 256, 0, st>>>(nS, 100, dS, 100, cnt,
        lo + L_N0_G, lo + L_N0_B, nS, 100, 100, nV, dV, nV, 16, N);
    // ff0: (100,16)->(400,32), relu + sigmoid; concat ld 160 (K=160)
    fill_kernel<<<l1d(N * 160), 256, 0, st>>>(nT2, 0.0f, N * 160);
    copy_block_kernel<<<l1d(N * 100), 256, 0, st>>>(nS, 100, nT2, 160, N, 100);
    vh_vn_kernel<<<l1d(N * 32), 256, 0, st>>>(nV, 16, lo + L_FF0_WH, 32, nVh, nT2, 160, 100, N);
    fill_kernel<<<l1d(N * 448), 256, 0, st>>>(nT1, 0.0f, N * 448);
    gemm(nT2, 160, wl + WL_FF0, bl + BL_FF0, nT1, 448, N, 400, 160, 1);
    wv_gate_kernel<<<l1d(N * 32), 256, 0, st>>>(nVh, 32, lo + L_FF0_WV, 32, nVt, 1, N);
    // ff1: (400,32)->(100,16), no acts; K=448
    vh_vn_kernel<<<l1d(N * 32), 256, 0, st>>>(nVt, 32, lo + L_FF1_WH, 32, nVh, nT1, 448, 400, N);
    gemm(nT1, 448, wl + WL_FF1, bl + BL_FF1, dS, 100, N, 100, 448, 0);
    wv_gate_kernel<<<l1d(N * 16), 256, 0, st>>>(nVh, 32, lo + L_FF1_WV, 16, dV, 0, N);
    // residual + norm1 (no cnt)
    add_ln_kernel<<<l1d(N), 256, 0, st>>>(nS, 100, dS, 100, nullptr,
        lo + L_N1_G, lo + L_N1_B, nS, 100, 100, nV, dV, nV, 16, N);
  }

  // ---- output: Wout_ln + Wout (100,16)->(100,), relu; concat ld 128 (K=128)
  add_ln_kernel<<<l1d(N), 256, 0, st>>>(nS, 100, nullptr, 0, nullptr,
      prm + P_WOUTLN_G, prm + P_WOUTLN_B, nS, 100, 100, nV, nullptr, nV, 16, N);
  fill_kernel<<<l1d(N * 128), 256, 0, st>>>(nT2, 0.0f, N * 128);
  copy_block_kernel<<<l1d(N * 100), 256, 0, st>>>(nS, 100, nT2, 128, N, 100);
  vh_vn_kernel<<<l1d(N * 16), 256, 0, st>>>(nV, 16, prm + P_WOUT_WH, 16, nVh, nT2, 128, 100, N);
  gemm(nT2, 128, wp + WP_WOUT, bp + BP_WOUT, dS, 100, N, 100, 128, 1);

  // ---- graph mean pooling
  scatter_add_kernel<<<l1d(N * 100), 256, 0, st>>>(dS, 100, 100, batch, pooled, N);
  count_kernel<<<l1d(N), 256, 0, st>>>(batch, gcnt, N);
}

// ---------------- entry -----------------------------------------------------
extern "C" void kernel_launch(void* const* d_in, const int* in_sizes, int n_in,
                              void* d_out, int out_size, void* d_ws, size_t ws_size,
                              hipStream_t stream) {
  const float* hVp_s = (const float*)d_in[0];
  const float* hVp_v = (const float*)d_in[1];
  const int*   eip   = (const int*)d_in[2];
  const float* hEp_s = (const float*)d_in[3];
  const float* hEp_v = (const float*)d_in[4];
  const int*   bpar  = (const int*)d_in[5];
  const float* hVs_s = (const float*)d_in[6];
  const float* hVs_v = (const float*)d_in[7];
  const int*   eis   = (const int*)d_in[8];
  const float* hEs_s = (const float*)d_in[9];
  const float* hEs_v = (const float*)d_in[10];
  const int*   bsub  = (const int*)d_in[11];
  const int*   labels= (const int*)d_in[12];
  const float* prm   = (const float*)d_in[13];

  const long N_PAR = 4096, E_PAR = 65536;
  const long N_SUB = 1024, E_SUB = 16384;

  float* ws = (float*)d_ws;
  float* pooledP = ws;                 // 16*100
  float* gcntP   = ws + 1600;          // 16
  float* pooledS = ws + 1664;          // 16*100
  float* gcntS   = ws + 3264;          // 16
  float* biasPad = ws + 3328;          // BP_TOTAL floats (3648)
  f16*   wpack   = (f16*)(ws + 7040);  // WP_TOTAL halfs (720384 -> 360192 floats)
  float* arena   = ws + 7040 + 360192; // ~22M floats reused by both encodes

  // ---- pack all projection weights to padded f16 (+ padded f32 bias) ------
  struct PackD { long w, b; int Nout, K, Np, Kp; long wpo, bpo; };
  PackD pk[23];
  int np = 0;
  pk[np++] = {P_WV_WSW,   P_WV_WSB,   100,  22, 112,  32, WP_WV,   BP_WV};
  pk[np++] = {P_WE_WSW,   P_WE_WSB,    32,  33,  32,  64, WP_WE,   BP_WE};
  pk[np++] = {P_WOUT_WSW, P_WOUT_WSB, 100, 116, 112, 128, WP_WOUT, BP_WOUT};
  for (int l = 0; l < 4; ++l) {
    long lp = P_L0 + (long)l * P_LSTRIDE;
    long wl = WP_L0 + (long)l * WP_LSTRIDE;
    long bl = BP_L0 + (long)l * BP_LSTRIDE;
    pk[np++] = {lp + L_M0_WSW,  lp + L_M0_WSB,  100, 265, 112, 288, wl + WL_M0,  bl + BL_M0};
    pk[np++] = {lp + L_M1_WSW,  lp + L_M1_WSB,  100, 116, 112, 128, wl + WL_M1,  bl + BL_M1};
    pk[np++] = {lp + L_M2_WSW,  lp + L_M2_WSB,  100, 116, 112, 128, wl + WL_M2,  bl + BL_M2};
    pk[np++] = {lp + L_FF0_WSW, lp + L_FF0_WSB, 400, 132, 400, 160, wl + WL_FF0, bl + BL_FF0};
    pk[np++] = {lp + L_FF1_WSW, lp + L_FF1_WSB, 100, 432, 112, 448, wl + WL_FF1, bl + BL_FF1};
  }
  for (int i = 0; i < np; ++i) {
    long tot = (long)pk[i].Np * pk[i].Kp;
    pack_w_kernel<<<dim3((unsigned)cdiv(tot, 256)), 256, 0, stream>>>(
        prm + pk[i].w, pk[i].Nout, pk[i].K, wpack + pk[i].wpo, pk[i].Kp, tot);
    pack_b_kernel<<<dim3((unsigned)cdiv(pk[i].Np, 256)), 256, 0, stream>>>(
        prm + pk[i].b, pk[i].Nout, biasPad + pk[i].bpo, pk[i].Np);
  }

  encode(hVp_s, hVp_v, eip, hEp_s, hEp_v, bpar, N_PAR, E_PAR, prm,
         wpack, biasPad, arena, pooledP, gcntP, stream);
  encode(hVs_s, hVs_v, eis, hEs_s, hEs_v, bsub, N_SUB, E_SUB, prm,
         wpack, biasPad, arena, pooledS, gcntS, stream);

  head_kernel<<<dim3(1), dim3(32), 0, stream>>>(pooledP, gcntP, pooledS, gcntS,
                                                prm, labels, (float*)d_out);
}
